// ELMO_51848845197906
// MI455X (gfx1250) — compile-verified, matrix operand order checked by reading, not run on hardware
//
#include <hip/hip_runtime.h>
#include <hip/hip_bf16.h>

// ELMo-style biLM forward for MI455X (gfx1250, wave32, WMMA).
// Phases: (1) embed gather -> f16, (2) pack all weights f32 -> f16 in WMMA
// B-fragment order, (3) persistent 2-layer LSTM per direction (1 WGP each,
// 32 waves, v_wmma_f32_16x16x32_f16 gate GEMMs, f32 cell state in registers,
// f16 hidden state in LDS), (4) big projection GEMM [2016x1024]x[1024x32000]
// with f16 WMMA + f32 accumulate + bias.  B-fragments use a 2-stage unrolled
// ping-pong pipeline (no register copies -> no WMMA->VALU hazard NOPs) so
// chunk c+1's global_load_b128s overlap chunk c's WMMAs.
//
// Requires ws_size >= ~85 MB.

#define V_  32000
#define D_  256
#define U_  512
#define B_  16
#define T_  128
#define G_  2048            // 4*U
#define M_ROWS 2016         // B_*126
#define KP 1024             // 2*U (projection K)

typedef __attribute__((ext_vector_type(16))) _Float16 v16h;
typedef __attribute__((ext_vector_type(8)))  _Float16 v8h;
typedef __attribute__((ext_vector_type(4)))  _Float16 v4h;
typedef __attribute__((ext_vector_type(8)))  float    v8f;

static __device__ __forceinline__ float sigm(float x) {
    return 1.0f / (1.0f + __expf(-x));
}

// K-index within a 32-wide chunk for f16 A/B WMMA fragments (ISA 7.12.2):
// elements 0..7 -> (lane/16)*8 + e ; elements 8..15 -> 16 + (lane/16)*8 + (e-8)
static __device__ __forceinline__ int kmap(int e, int lane) {
    int hi = lane >> 4;
    return (e < 8) ? (hi * 8 + e) : (16 + hi * 8 + (e - 8));
}

// Build an A (or B) fragment from a row-major f16 tile in LDS/global.
// base points at [16 x stride] matrix; rows indexed by lane%16.
static __device__ __forceinline__ v16h frag_from_rowmajor(const _Float16* base,
                                                          int stride, int lane, int k0) {
    const int m  = lane & 15;
    const int hi = lane >> 4;
    const _Float16* p = base + m * stride + k0 + hi * 8;
    v8h lo  = *(const v8h*)p;          // K = k0 + hi*8 + 0..7  (16B contiguous)
    v8h hi8 = *(const v8h*)(p + 16);   // K = k0 + 16 + hi*8 + 0..7
    v16h r;
#pragma unroll
    for (int i = 0; i < 8; ++i) { r[i] = lo[i]; r[i + 8] = hi8[i]; }
    return r;
}

// ---------------------------------------------------------------------------
// Phase 1: embedding gather + f32 -> f16
// ---------------------------------------------------------------------------
__global__ void embed_gather_kernel(const int* __restrict__ seqs,
                                    const float* __restrict__ embed,
                                    _Float16* __restrict__ emb16) {
    const int idx = blockIdx.x * 256 + threadIdx.x;   // exactly B_*T_*D_ threads
    const int d  = idx & (D_ - 1);
    const int bt = idx >> 8;                           // b*T_ + t
    const int tok = seqs[bt];
    emb16[idx] = (_Float16)embed[(long long)tok * D_ + d];
}

// ---------------------------------------------------------------------------
// Phase 2: pack a f32 [K x N] weight into f16 WMMA B-fragment order.
// dst element idx = ((ktile*(N/16)+ntile)*32 + lane)*16 + e  (coalesced writes)
// ---------------------------------------------------------------------------
__global__ void pack_weight_kernel(const float* __restrict__ src,
                                   _Float16* __restrict__ dst,
                                   int K, int N) {
    const long long idx = (long long)blockIdx.x * 256 + threadIdx.x;
    const long long total = (long long)K * N;
    if (idx >= total) return;
    const int e    = (int)(idx & 15);
    const int lane = (int)((idx >> 4) & 31);
    const long long f = idx >> 9;
    const int NT = N >> 4;
    const int kt = (int)(f / NT);
    const int nt = (int)(f % NT);
    const int k = kt * 32 + kmap(e, lane);
    const int n = nt * 16 + (lane & 15);
    dst[idx] = (_Float16)src[(long long)k * N + n];
}

// ---------------------------------------------------------------------------
// Gate GEMM: z[16 x 2048] = a0[16 x k0len]@W0 + a1[16 x k1len]@W1 + bias
// 32 waves x 4 N-tiles each; f32 accumulate; 2-stage ping-pong B pipeline.
// Chunk count (k0len+k1len)/32 is even (24 or 32).
// ---------------------------------------------------------------------------
static __device__ __forceinline__ void gate_gemm(
        const _Float16* __restrict__ a0, int k0len, const _Float16* __restrict__ w0,
        const _Float16* __restrict__ a1, int k1len, const _Float16* __restrict__ w1,
        const float* __restrict__ bias, float* __restrict__ z,
        int lane, int wave) {
    const int NT = G_ / 16;   // 128 N-tiles
    const int nl = lane & 15;
    const int hi = lane >> 4;
    const int c0 = k0len >> 5;
    const int ct = c0 + (k1len >> 5);

    v8f acc[4];
#pragma unroll
    for (int j = 0; j < 4; ++j) {
        const float bv = bias[(wave * 4 + j) * 16 + nl];
#pragma unroll
        for (int r = 0; r < 8; ++r) acc[j][r] = bv;
    }

    auto load_a = [&](int c) -> v16h {
        return (c < c0) ? frag_from_rowmajor(a0, k0len, lane, c * 32)
                        : frag_from_rowmajor(a1, k1len, lane, (c - c0) * 32);
    };
    auto load_b = [&](int c, int j) -> v16h {
        const _Float16* w = (c < c0) ? w0 : w1;
        const int kt = (c < c0) ? c : (c - c0);
        const int nt = wave * 4 + j;
        return *(const v16h*)(w + ((long long)(kt * NT + nt) << 9) + lane * 16);
    };

    v16h bfA[4], bfB[4];
#pragma unroll
    for (int j = 0; j < 4; ++j) bfA[j] = load_b(0, j);

    for (int c = 0; c < ct; c += 2) {
#pragma unroll
        for (int j = 0; j < 4; ++j) bfB[j] = load_b(c + 1, j);
        {
            const v16h af = load_a(c);
#pragma unroll
            for (int j = 0; j < 4; ++j)
                acc[j] = __builtin_amdgcn_wmma_f32_16x16x32_f16(
                    false, af, false, bfA[j], (short)0, acc[j], false, false);
        }
        if (c + 2 < ct) {
#pragma unroll
            for (int j = 0; j < 4; ++j) bfA[j] = load_b(c + 2, j);
        }
        {
            const v16h af = load_a(c + 1);
#pragma unroll
            for (int j = 0; j < 4; ++j)
                acc[j] = __builtin_amdgcn_wmma_f32_16x16x32_f16(
                    false, af, false, bfB[j], (short)0, acc[j], false, false);
        }
    }

#pragma unroll
    for (int j = 0; j < 4; ++j) {
        const int col = (wave * 4 + j) * 16 + nl;
#pragma unroll
        for (int r = 0; r < 8; ++r)
            z[(r + 8 * hi) * G_ + col] = acc[j][r];
    }
}

// ---------------------------------------------------------------------------
// Phase 3: persistent 2-layer LSTM, one workgroup per direction.
// blockIdx.x: 0 = forward, 1 = backward.  Dynamic LDS: 168 KB.
//   z   : 16 x 2048 f32 (131072 B)
//   x   : 16 x 256  f16 (  8192 B)
//   h0  : 16 x 512  f16 ( 16384 B)
//   h1  : 16 x 512  f16 ( 16384 B)
// ---------------------------------------------------------------------------
#define LSTM_SMEM (131072 + 8192 + 16384 + 16384)

__global__ void __launch_bounds__(1024, 1)
lstm_kernel(const int* __restrict__ seqs, const _Float16* __restrict__ emb,
            const _Float16* fW0, const _Float16* fU0, const float* fb0,
            const _Float16* fW1, const _Float16* fU1, const float* fb1,
            const _Float16* bW0, const _Float16* bU0, const float* bb0,
            const _Float16* bW1, const _Float16* bU1, const float* bb1,
            _Float16* __restrict__ o_out) {
    extern __shared__ __align__(16) char smem[];
    float*    z     = (float*)smem;
    _Float16* x_lds = (_Float16*)(smem + 131072);
    _Float16* h0    = (_Float16*)(smem + 131072 + 8192);
    _Float16* h1    = (_Float16*)(smem + 131072 + 8192 + 16384);

    const int  tid  = threadIdx.x;
    const int  lane = tid & 31;
    const int  wave = tid >> 5;
    const bool fwd  = (blockIdx.x == 0);

    const _Float16* W0 = fwd ? fW0 : bW0;
    const _Float16* U0 = fwd ? fU0 : bU0;
    const _Float16* W1 = fwd ? fW1 : bW1;
    const _Float16* U1 = fwd ? fU1 : bU1;
    const float*    B0 = fwd ? fb0 : bb0;
    const float*    B1 = fwd ? fb1 : bb1;

    // Elementwise ownership: 1024 threads x 8 units = 16 batch x 512 units.
    const int eb  = tid >> 6;          // batch row 0..15
    const int eu0 = (tid & 63) * 8;    // first hidden unit

    for (int i = tid; i < B_ * U_; i += 1024) {
        h0[i] = (_Float16)0.0f;
        h1[i] = (_Float16)0.0f;
    }
    float c0[8], c1[8];
#pragma unroll
    for (int r = 0; r < 8; ++r) { c0[r] = 0.0f; c1[r] = 0.0f; }
    __syncthreads();

    for (int s = 0; s < T_ - 1; ++s) {
        const int t = fwd ? s : (T_ - 1 - s);

        // stage x_t (16 x 256 f16) into LDS: 4 halfs per thread
        {
            const int i = tid * 4;
            const int b = i >> 8, d = i & (D_ - 1);
            *(v4h*)&x_lds[i] = *(const v4h*)&emb[((b * T_ + t) << 8) + d];
        }
        __syncthreads();

        // ---- layer 0 gates ----
        gate_gemm(x_lds, D_, W0, h0, U_, U0, B0, z, lane, wave);
        __syncthreads();

        const bool m = (seqs[eb * T_ + t] != 0);
#pragma unroll
        for (int r = 0; r < 8; ++r) {
            const int u = eu0 + r;
            const float iv = z[eb * G_ + u];
            const float fv = z[eb * G_ + U_ + u];
            const float gv = z[eb * G_ + 2 * U_ + u];
            const float ov = z[eb * G_ + 3 * U_ + u];
            float cn = sigm(fv) * c0[r] + sigm(iv) * tanhf(gv);
            float hn = sigm(ov) * tanhf(cn);
            if (!m) { hn = (float)h0[eb * U_ + u]; cn = c0[r]; }
            c0[r] = cn;
            h0[eb * U_ + u] = (_Float16)hn;
        }
        __syncthreads();

        // ---- layer 1 gates ----
        gate_gemm(h0, U_, W1, h1, U_, U1, B1, z, lane, wave);
        __syncthreads();

        const int orow = eb * 126 + (fwd ? s : (125 - s));
        const int ocol = fwd ? 0 : U_;
#pragma unroll
        for (int r = 0; r < 8; ++r) {
            const int u = eu0 + r;
            const float iv = z[eb * G_ + u];
            const float fv = z[eb * G_ + U_ + u];
            const float gv = z[eb * G_ + 2 * U_ + u];
            const float ov = z[eb * G_ + 3 * U_ + u];
            float cn = sigm(fv) * c1[r] + sigm(iv) * tanhf(gv);
            float hn = sigm(ov) * tanhf(cn);
            if (!m) { hn = (float)h1[eb * U_ + u]; cn = c1[r]; }
            c1[r] = cn;
            const _Float16 hh = (_Float16)hn;
            h1[eb * U_ + u] = hh;
            if (s <= 125) o_out[(long long)orow * KP + ocol + u] = hh;
        }
        __syncthreads();
    }
}

// ---------------------------------------------------------------------------
// Phase 4: projection GEMM: out[2016 x 32000] = o16[2016 x 1024] @ Wp + bp.
// Block: 128 threads (4 waves), each wave 4 N-tiles -> 16x256 f32 per block.
// Grid: (2000/16, 2016/16) = (125, 126).  2-stage ping-pong B pipeline.
// ---------------------------------------------------------------------------
__global__ void __launch_bounds__(128)
proj_kernel(const _Float16* __restrict__ o16, const _Float16* __restrict__ wp,
            const float* __restrict__ bp, float* __restrict__ out) {
    __shared__ _Float16 a_lds[16 * KP];   // 32 KB A tile
    const int tid  = threadIdx.x;
    const int lane = tid & 31;
    const int wave = tid >> 5;
    const int mt   = blockIdx.y;          // M tile 0..125
    const int ntb  = blockIdx.x;          // group of 16 N-tiles

    // Stage contiguous A tile (rows mt*16 .. mt*16+15)
    for (int i = tid * 8; i < 16 * KP; i += 128 * 8)
        *(v8h*)&a_lds[i] = *(const v8h*)&o16[(long long)mt * 16 * KP + i];
    __syncthreads();

    const int NT = V_ / 16;               // 2000
    const int nl = lane & 15;
    const int hi = lane >> 4;
    v8f acc[4];
#pragma unroll
    for (int j = 0; j < 4; ++j) {
        const float bv = bp[ntb * 256 + (wave * 4 + j) * 16 + nl];
#pragma unroll
        for (int r = 0; r < 8; ++r) acc[j][r] = bv;
    }

    auto load_b = [&](int kt, int j) -> v16h {
        const int nt = ntb * 16 + wave * 4 + j;
        return *(const v16h*)(wp + ((long long)(kt * NT + nt) << 9) + lane * 16);
    };

    v16h bfA[4], bfB[4];
#pragma unroll
    for (int j = 0; j < 4; ++j) bfA[j] = load_b(0, j);

    const int KT = KP / 32;               // 32 K-chunks (even)
    for (int kt = 0; kt < KT; kt += 2) {
#pragma unroll
        for (int j = 0; j < 4; ++j) bfB[j] = load_b(kt + 1, j);
        {
            const v16h af = frag_from_rowmajor(a_lds, KP, lane, kt * 32);
#pragma unroll
            for (int j = 0; j < 4; ++j)
                acc[j] = __builtin_amdgcn_wmma_f32_16x16x32_f16(
                    false, af, false, bfA[j], (short)0, acc[j], false, false);
        }
        if (kt + 2 < KT) {
#pragma unroll
            for (int j = 0; j < 4; ++j) bfA[j] = load_b(kt + 2, j);
        }
        {
            const v16h af = frag_from_rowmajor(a_lds, KP, lane, (kt + 1) * 32);
#pragma unroll
            for (int j = 0; j < 4; ++j)
                acc[j] = __builtin_amdgcn_wmma_f32_16x16x32_f16(
                    false, af, false, bfB[j], (short)0, acc[j], false, false);
        }
    }

#pragma unroll
    for (int j = 0; j < 4; ++j) {
        const int n = ntb * 256 + (wave * 4 + j) * 16 + nl;
#pragma unroll
        for (int r = 0; r < 8; ++r) {
            const int row = mt * 16 + r + 8 * hi;
            out[(long long)row * V_ + n] = acc[j][r];
        }
    }
}

// ---------------------------------------------------------------------------
extern "C" void kernel_launch(void* const* d_in, const int* in_sizes, int n_in,
                              void* d_out, int out_size, void* d_ws, size_t ws_size,
                              hipStream_t stream) {
    (void)in_sizes; (void)n_in; (void)out_size; (void)ws_size;
    const int*   seqs  = (const int*)d_in[0];
    const float* embed = (const float*)d_in[1];
    const float* fW0 = (const float*)d_in[2];
    const float* fU0 = (const float*)d_in[3];
    const float* fb0 = (const float*)d_in[4];
    const float* fW1 = (const float*)d_in[5];
    const float* fU1 = (const float*)d_in[6];
    const float* fb1 = (const float*)d_in[7];
    const float* bW0 = (const float*)d_in[8];
    const float* bU0 = (const float*)d_in[9];
    const float* bb0 = (const float*)d_in[10];
    const float* bW1 = (const float*)d_in[11];
    const float* bU1 = (const float*)d_in[12];
    const float* bb1 = (const float*)d_in[13];
    const float* Wp  = (const float*)d_in[14];
    const float* bp  = (const float*)d_in[15];

    char* ws = (char*)d_ws;
    size_t off = 0;
    auto take = [&](size_t bytes) {
        size_t o = off;
        off += (bytes + 255) & ~(size_t)255;
        return (void*)(ws + o);
    };
    _Float16* emb16 = (_Float16*)take((size_t)B_ * T_ * D_ * 2);
    _Float16* fW0p  = (_Float16*)take((size_t)D_ * G_ * 2);
    _Float16* fU0p  = (_Float16*)take((size_t)U_ * G_ * 2);
    _Float16* fW1p  = (_Float16*)take((size_t)U_ * G_ * 2);
    _Float16* fU1p  = (_Float16*)take((size_t)U_ * G_ * 2);
    _Float16* bW0p  = (_Float16*)take((size_t)D_ * G_ * 2);
    _Float16* bU0p  = (_Float16*)take((size_t)U_ * G_ * 2);
    _Float16* bW1p  = (_Float16*)take((size_t)U_ * G_ * 2);
    _Float16* bU1p  = (_Float16*)take((size_t)U_ * G_ * 2);
    _Float16* Wpp   = (_Float16*)take((size_t)KP * V_ * 2);
    _Float16* o16   = (_Float16*)take((size_t)M_ROWS * KP * 2);

    // Phase 1: embedding gather
    embed_gather_kernel<<<(B_ * T_ * D_) / 256, 256, 0, stream>>>(seqs, embed, emb16);

    // Phase 2: pack all weights into WMMA fragment order (f16)
    auto pack = [&](const float* src, _Float16* dst, int K, int N) {
        const long long total = (long long)K * N;
        pack_weight_kernel<<<(unsigned)((total + 255) / 256), 256, 0, stream>>>(src, dst, K, N);
    };
    pack(fW0, fW0p, D_, G_);  pack(fU0, fU0p, U_, G_);
    pack(fW1, fW1p, U_, G_);  pack(fU1, fU1p, U_, G_);
    pack(bW0, bW0p, D_, G_);  pack(bU0, bU0p, U_, G_);
    pack(bW1, bW1p, U_, G_);  pack(bU1, bU1p, U_, G_);
    pack(Wp, Wpp, KP, V_);

    // Phase 3: persistent biLSTM (one WGP per direction, 168 KB dynamic LDS)
    hipFuncSetAttribute((const void*)lstm_kernel,
                        hipFuncAttributeMaxDynamicSharedMemorySize, LSTM_SMEM);
    lstm_kernel<<<2, 1024, LSTM_SMEM, stream>>>(
        seqs, emb16,
        fW0p, fU0p, fb0, fW1p, fU1p, fb1,
        bW0p, bU0p, bb0, bW1p, bU1p, bb1,
        o16);

    // Phase 4: projection GEMM + bias -> d_out (f32)
    proj_kernel<<<dim3(V_ / 256, M_ROWS / 16), 128, 0, stream>>>(
        o16, Wpp, bp, (float*)d_out);
}
